// GPTStyleTransformer_15058155340489
// MI455X (gfx1250) — matrix operand order repaired and checked by reading, not compiled, not run on hardware
//
#include <hip/hip_runtime.h>
#include <hip/hip_bf16.h>
#include <stdint.h>

typedef __attribute__((ext_vector_type(16))) _Float16 v16h;
typedef __attribute__((ext_vector_type(8)))  _Float16 v8h;
typedef __attribute__((ext_vector_type(4)))  _Float16 v4h;
typedef __attribute__((ext_vector_type(8)))  float    v8f;
typedef __attribute__((ext_vector_type(4)))  unsigned int v4u;
typedef __attribute__((ext_vector_type(8)))  int      v8i;
typedef __attribute__((ext_vector_type(4)))  int      v4i;

#define TBLK 256
#define BM 64
#define BN 64
#define BK 64
#define LDST 72   // LDS tile row stride in halves: 64 data + 8 pad

#if __has_builtin(__builtin_amdgcn_tensor_load_to_lds) && __has_builtin(__builtin_amdgcn_s_wait_tensorcnt)
#define USE_TDM 1
#else
#define USE_TDM 0
#endif

__device__ __forceinline__ v16h cat8(v8h lo, v8h hi)
{
    v16h r;
#pragma unroll
    for (int e = 0; e < 8; ++e) { r[e] = lo[e]; r[e + 8] = hi[e]; }
    return r;
}

#if USE_TDM
// ---------------------------------------------------------------------------
// Issue one TDM 2D tile load: rows_valid x 64 f16 elements, row stride
// ld_elems, from gsrc into LDS at byte offset lds_off, padded +8 halves/row
// (pad_interval 32 DWORDs -> code 4, pad_amount 4 DWORDs -> code 3).
// D# per ISA 8.3-8.5; groups 2/3 zero (2D tensor). Rows beyond rows_valid
// read as zero (OOB fill) -> handles the lm_head N tail.
// This toolchain exposes the 6-arg builtin form:
//   (uint32x4 g0, int32x8 g1, int32x4, int32x4, int32x8, i32 cpol)
// ---------------------------------------------------------------------------
__device__ __forceinline__ void tdm_load_tile_f16(uint32_t lds_off,
                                                  const _Float16* gsrc,
                                                  uint32_t ld_elems,
                                                  uint32_t rows_valid)
{
    const uint64_t ga = (uint64_t)(uintptr_t)gsrc;
    v4u g0;
    g0[0] = 1u;                                   // count=1, user descriptor
    g0[1] = lds_off;                              // lds_addr
    g0[2] = (uint32_t)(ga & 0xFFFFFFFFu);         // global_addr lo
    g0[3] = (uint32_t)(ga >> 32) | (2u << 30);    // global_addr hi | type=2
    const uint32_t td0 = 64u;                     // tensor_dim0 (cols)
    const uint32_t td1 = rows_valid;              // tensor_dim1 (rows)
    v8i g1;
    g1[0] = (int)((1u << 16) |                    // data_size = 2 bytes
                  (1u << 20) |                    // pad_enable
                  (4u << 22) |                    // pad_interval = 32 DWORDs
                  (3u << 25));                    // pad_amount   = 4 DWORDs
    g1[1] = (int)((td0 & 0xFFFFu) << 16);
    g1[2] = (int)((td0 >> 16) | ((td1 & 0xFFFFu) << 16));
    g1[3] = (int)((td1 >> 16) | (64u << 16));     // tile_dim0 = 64
    g1[4] = (int)64u;                             // tile_dim1 = 64, tile_dim2 = 0
    g1[5] = (int)(ld_elems);                      // tensor_dim0_stride lo32
    g1[6] = 0;                                    // stride hi16 | dim1_stride lo
    g1[7] = 0;
    const v4i gz4 = {0, 0, 0, 0};
    const v8i gz8 = {0, 0, 0, 0, 0, 0, 0, 0};
    __builtin_amdgcn_tensor_load_to_lds(g0, g1, gz4, gz4, gz8, 0);
}
#endif

// ---------------------------------------------------------------------------
// WMMA GEMM, f16 inputs, f32 accumulate:
//   C[M,N] = alpha * A[M,K] x Bop + bias[N]
//   TRANSB=1: Bop = B^T, B stored [N,K];  TRANSB=0: Bop = B stored [K,N]
//   OUTH: store C as f16 (activations feeding further GEMMs) else f32.
// Assumes M%64==0, K%64==0. NGUARD only for lm_head (N=50257).
// Block: 8 wave32 in 4(m)x2(n) grid; each wave: 16x32 of C (2 accumulators).
// LDS tiles are plain row-major (row stride 72 halves); fragments are flat
// 16B ds loads per ISA 7.12.2. Staging = raw 2D f16 tile copy:
// TDM (tensor_load_to_lds) double-buffered, or vectorized fallback.
// ---------------------------------------------------------------------------
template <int TRANSB, bool NGUARD, bool OUTH>
__global__ __launch_bounds__(TBLK)
void k_gemm_wmma(const _Float16* __restrict__ A, int lda,
                 const _Float16* __restrict__ B, int ldb,
                 void* __restrict__ Cv, int ldc,
                 int M, int N, int K,
                 float alpha, const float* __restrict__ bias,
                 int causal_skip)
{
    __shared__ _Float16 At[2][BM][LDST];
    __shared__ _Float16 Bt[2][BK][LDST];

    const int tid  = threadIdx.x;
    const int lane = tid & 31;
    const int wave = tid >> 5;
    const int wr   = wave >> 1;          // 0..3 m-subtile
    const int wc   = wave & 1;           // 0..1 n pair
    const int mBlk = blockIdx.y * BM;
    const int nBlk = blockIdx.x * BN;

    if (causal_skip && nBlk > mBlk + BM - 1) return;   // fully-masked tile

    v8f acc0 = {};
    v8f acc1 = {};

    const int l15 = lane & 15;
    const int lhi = lane >> 4;

    // rows of the B tile that exist (TDM zero-fills the rest; fallback guards)
    const int brows = NGUARD ? (N - nBlk < BK ? (N - nBlk) : BK) : BK;

#if USE_TDM
    const bool issuer = (wave == 0);
    if (issuer) {
        tdm_load_tile_f16((uint32_t)(uintptr_t)&At[0][0][0],
                          A + (size_t)mBlk * lda, (uint32_t)lda, BM);
        const _Float16* bsrc = TRANSB ? (B + (size_t)nBlk * ldb) : (B + nBlk);
        tdm_load_tile_f16((uint32_t)(uintptr_t)&Bt[0][0][0],
                          bsrc, (uint32_t)ldb, (uint32_t)(TRANSB ? brows : BK));
    }
    int p = 0;
    for (int k0 = 0; k0 < K; k0 += BK) {
        if (issuer) __builtin_amdgcn_s_wait_tensorcnt((short)0);
        __syncthreads();
        if (issuer && (k0 + BK < K)) {
            tdm_load_tile_f16((uint32_t)(uintptr_t)&At[p ^ 1][0][0],
                              A + (size_t)mBlk * lda + (k0 + BK),
                              (uint32_t)lda, BM);
            const _Float16* bsrc = TRANSB
                ? (B + (size_t)nBlk * ldb + (k0 + BK))
                : (B + (size_t)(k0 + BK) * ldb + nBlk);
            tdm_load_tile_f16((uint32_t)(uintptr_t)&Bt[p ^ 1][0][0],
                              bsrc, (uint32_t)ldb,
                              (uint32_t)(TRANSB ? brows : BK));
        }
#else
    const int p = 0;
    for (int k0 = 0; k0 < K; k0 += BK) {
        // ---- fallback staging: batched v8h copies, then one wait ----
        v8h ra[2], rb[2];
#pragma unroll
        for (int j = 0; j < 2; ++j) {
            const int t = tid + j * TBLK;        // 0..511: 64 rows x 8 chunks
            const int r = t >> 3;
            const int c = (t & 7) * 8;
            ra[j] = *(const v8h*)&A[(size_t)(mBlk + r) * lda + (k0 + c)];
            v8h bz = {};
            if (TRANSB) {
                if (!NGUARD || r < brows)
                    bz = *(const v8h*)&B[(size_t)(nBlk + r) * ldb + (k0 + c)];
            } else {
                bz = *(const v8h*)&B[(size_t)(k0 + r) * ldb + (nBlk + c)];
            }
            rb[j] = bz;
        }
#pragma unroll
        for (int j = 0; j < 2; ++j) {
            const int t = tid + j * TBLK;
            const int r = t >> 3;
            const int c = (t & 7) * 8;
            *(v8h*)&At[0][r][c] = ra[j];
            *(v8h*)&Bt[0][r][c] = rb[j];
        }
        __syncthreads();
#endif
        // ---- 4 WMMAs per stage: 2 k-steps x 2 n-subtiles, shared A frag ----
#pragma unroll
        for (int ks = 0; ks < 2; ++ks) {
            const int arow = wr * 16 + l15;
            const int kb   = 16 * lhi;           // A: lanes 0-15 k0..15
            const v8h alo = *(const v8h*)&At[p][arow][ks * 32 + kb];
            const v8h ahi = *(const v8h*)&At[p][arow][ks * 32 + kb + 8];
            const v16h af = cat8(alo, ahi);
#pragma unroll
            for (int nsub = 0; nsub < 2; ++nsub) {
                v16h bf;
                if (TRANSB) {
                    const int brow = wc * 32 + nsub * 16 + l15;
                    const int kb2  = 16 * lhi;   // B: lanes 0-15 k0..15
                    const v8h blo = *(const v8h*)&Bt[p][brow][ks * 32 + kb2];
                    const v8h bhi = *(const v8h*)&Bt[p][brow][ks * 32 + kb2 + 8];
                    bf = cat8(blo, bhi);
                } else {
                    const int ncol = wc * 32 + nsub * 16 + l15;
                    const int krow = ks * 32 + 16 * lhi;
#pragma unroll
                    for (int e = 0; e < 16; ++e) bf[e] = Bt[p][krow + e][ncol];
                }
                if (nsub == 0)
                    acc0 = __builtin_amdgcn_wmma_f32_16x16x32_f16(
                               false, af, false, bf, (short)0, acc0, false, false);
                else
                    acc1 = __builtin_amdgcn_wmma_f32_16x16x32_f16(
                               false, af, false, bf, (short)0, acc1, false, false);
            }
        }
        __syncthreads();
#if USE_TDM
        p ^= 1;
#endif
    }

    // ---- epilogue: C layout — VGPR r, lane l: M = r + 8*l[4], N = l&15 ----
#pragma unroll
    for (int nsub = 0; nsub < 2; ++nsub) {
        const int col = nBlk + wc * 32 + nsub * 16 + l15;
        float bv = 0.f;
        if (bias != nullptr && (!NGUARD || col < N)) bv = bias[col];
        const v8f* accp = nsub == 0 ? &acc0 : &acc1;
#pragma unroll
        for (int r = 0; r < 8; ++r) {
            const int grow = mBlk + wr * 16 + r + 8 * lhi;
            if (!NGUARD || col < N) {
                const float val = alpha * (*accp)[r] + bv;
                if (OUTH) ((_Float16*)Cv)[(size_t)grow * ldc + col] = (_Float16)val;
                else      ((float*)Cv)[(size_t)grow * ldc + col] = val;
            }
        }
    }
}

// ---------------------------------------------------------------------------
// fp32 -> fp16 bulk convert (n multiple of 4)
// ---------------------------------------------------------------------------
__global__ void k_cvt_f16(const float* __restrict__ src, _Float16* __restrict__ dst,
                          int n4)
{
    const int i = blockIdx.x * blockDim.x + threadIdx.x;
    if (i < n4) {
        const float4 f = ((const float4*)src)[i];
        v4h h = { (_Float16)f.x, (_Float16)f.y, (_Float16)f.z, (_Float16)f.w };
        ((v4h*)dst)[i] = h;
    }
}

// ---------------------------------------------------------------------------
// x[t,:] = emb[tok[t],:] + sinusoidal PE(t,:)   (fp32 residual stream)
// ---------------------------------------------------------------------------
__global__ void k_embed(const int* __restrict__ tok, const float* __restrict__ emb,
                        float* __restrict__ x, int T, int D)
{
    const int t  = blockIdx.x;
    const int tk = tok[t];
    const float nlog = -logf(10000.0f) / (float)D;
    for (int d = threadIdx.x; d < D; d += blockDim.x) {
        const int i = d >> 1;
        const float freq = __expf(nlog * (float)(2 * i));
        const float ang  = (float)t * freq;
        const float pe   = (d & 1) ? __cosf(ang) : __sinf(ang);
        x[(size_t)t * D + d] = emb[(size_t)tk * D + d] + pe;
    }
}

// ---------------------------------------------------------------------------
// out_h[t,:] = f16( LN(x[t,:]) * g + b )
// ---------------------------------------------------------------------------
__global__ __launch_bounds__(TBLK)
void k_layernorm(const float* __restrict__ x, const float* __restrict__ g,
                 const float* __restrict__ b, _Float16* __restrict__ out, int D)
{
    const int t = blockIdx.x;
    const float* row = x + (size_t)t * D;
    __shared__ float red[TBLK];

    float s = 0.0f;
    for (int d = threadIdx.x; d < D; d += TBLK) s += row[d];
    red[threadIdx.x] = s;
    __syncthreads();
    for (int off = TBLK / 2; off > 0; off >>= 1) {
        if (threadIdx.x < off) red[threadIdx.x] += red[threadIdx.x + off];
        __syncthreads();
    }
    const float mean = red[0] / (float)D;
    __syncthreads();

    float s2 = 0.0f;
    for (int d = threadIdx.x; d < D; d += TBLK) {
        const float dv = row[d] - mean;
        s2 += dv * dv;
    }
    red[threadIdx.x] = s2;
    __syncthreads();
    for (int off = TBLK / 2; off > 0; off >>= 1) {
        if (threadIdx.x < off) red[threadIdx.x] += red[threadIdx.x + off];
        __syncthreads();
    }
    const float rstd = rsqrtf(red[0] / (float)D + 1e-5f);

    for (int d = threadIdx.x; d < D; d += TBLK) {
        out[(size_t)t * D + d] = (_Float16)((row[d] - mean) * rstd * g[d] + b[d]);
    }
}

// ---------------------------------------------------------------------------
// Causal softmax: read fp32 scores row t (valid j in [0,t]), write f16 probs
// with zeros for j > t so the probs x V GEMM runs dense.
// ---------------------------------------------------------------------------
__global__ __launch_bounds__(TBLK)
void k_softmax_causal(const float* __restrict__ s, _Float16* __restrict__ ph, int T)
{
    const int t = blockIdx.x;
    const float* row = s + (size_t)t * T;
    _Float16* prow   = ph + (size_t)t * T;
    const int n = t + 1;
    __shared__ float red[TBLK];

    float mx = -3.402823466e38f;
    for (int j = threadIdx.x; j < n; j += TBLK) mx = fmaxf(mx, row[j]);
    red[threadIdx.x] = mx;
    __syncthreads();
    for (int off = TBLK / 2; off > 0; off >>= 1) {
        if (threadIdx.x < off)
            red[threadIdx.x] = fmaxf(red[threadIdx.x], red[threadIdx.x + off]);
        __syncthreads();
    }
    mx = red[0];
    __syncthreads();

    float sum = 0.0f;
    for (int j = threadIdx.x; j < n; j += TBLK) sum += __expf(row[j] - mx);
    red[threadIdx.x] = sum;
    __syncthreads();
    for (int off = TBLK / 2; off > 0; off >>= 1) {
        if (threadIdx.x < off) red[threadIdx.x] += red[threadIdx.x + off];
        __syncthreads();
    }
    const float inv = 1.0f / red[0];

    for (int j = threadIdx.x; j < n; j += TBLK)
        prow[j] = (_Float16)(__expf(row[j] - mx) * inv);
    for (int j = n + threadIdx.x; j < T; j += TBLK)
        prow[j] = (_Float16)0.0f;
}

__global__ void k_add_inplace(float* __restrict__ x, const float* __restrict__ y, int n)
{
    const int i = blockIdx.x * blockDim.x + threadIdx.x;
    if (i < n) x[i] += y[i];
}

// exact gelu, fp32 in -> f16 out (feeds the W2 GEMM)
__global__ void k_gelu_h(const float* __restrict__ in, _Float16* __restrict__ out, int n)
{
    const int i = blockIdx.x * blockDim.x + threadIdx.x;
    if (i < n) {
        const float v = in[i];
        out[i] = (_Float16)(0.5f * v * (1.0f + erff(v * 0.70710678118654752f)));
    }
}

// ---------------------------------------------------------------------------
extern "C" void kernel_launch(void* const* d_in, const int* in_sizes, int n_in,
                              void* d_out, int out_size, void* d_ws, size_t ws_size,
                              hipStream_t stream)
{
    (void)in_sizes; (void)n_in; (void)out_size; (void)ws_size;

    const int L = 4, H = 12, HD = 64, D = 768, T = 2048, V = 50257;
    const int D4 = 4 * D;

    const int*   tok     = (const int*)  d_in[0];
    const float* emb     = (const float*)d_in[1];
    const float* lm_bias = (const float*)d_in[2];
    const float* ln1_g   = (const float*)d_in[3];
    const float* ln1_b   = (const float*)d_in[4];
    const float* ln2_g   = (const float*)d_in[5];
    const float* ln2_b   = (const float*)d_in[6];
    const float* Wq      = (const float*)d_in[7];
    const float* bq      = (const float*)d_in[8];
    const float* Wk      = (const float*)d_in[9];
    const float* bk      = (const float*)d_in[10];
    const float* Wv      = (const float*)d_in[11];
    const float* bv      = (const float*)d_in[12];
    const float* Wo      = (const float*)d_in[13];
    const float* bo      = (const float*)d_in[14];
    const float* W1      = (const float*)d_in[15];
    const float* b1      = (const float*)d_in[16];
    const float* W2      = (const float*)d_in[17];
    const float* b2      = (const float*)d_in[18];
    const float* lnf_g   = (const float*)d_in[19];
    const float* lnf_b   = (const float*)d_in[20];

    float* logits = (float*)d_out;                 // [T, V] fp32

    // ---- workspace carve-up (256B aligned) ----
    char* wsp = (char*)d_ws;
    auto alloc = [&](size_t bytes) -> char* {
        char* p = wsp;
        wsp += (bytes + 255) & ~(size_t)255;
        return p;
    };
    const size_t TD = (size_t)T * D;
    // fp32
    float* x      = (float*)alloc(TD * 4);             // residual stream
    float* tmp    = (float*)alloc(TD * 4);             // pre-residual proj out
    float* hbuf   = (float*)alloc((size_t)T * D4 * 4); // MLP hidden pre-gelu
    float* scores = (float*)alloc((size_t)T * T * 4);  // per-head scores
    // fp16 activations
    _Float16* xn_h   = (_Float16*)alloc(TD * 2);
    _Float16* qh     = (_Float16*)alloc(TD * 2);
    _Float16* kh     = (_Float16*)alloc(TD * 2);
    _Float16* vh     = (_Float16*)alloc(TD * 2);
    _Float16* ctx_h  = (_Float16*)alloc(TD * 2);
    _Float16* hh     = (_Float16*)alloc((size_t)T * D4 * 2);
    _Float16* prob_h = (_Float16*)alloc((size_t)T * T * 2);
    // fp16 weights (converted once per call)
    _Float16* emb_h = (_Float16*)alloc((size_t)V * D * 2);
    _Float16* Wq_h  = (_Float16*)alloc((size_t)L * D * D * 2);
    _Float16* Wk_h  = (_Float16*)alloc((size_t)L * D * D * 2);
    _Float16* Wv_h  = (_Float16*)alloc((size_t)L * D * D * 2);
    _Float16* Wo_h  = (_Float16*)alloc((size_t)L * D * D * 2);
    _Float16* W1_h  = (_Float16*)alloc((size_t)L * D4 * D * 2);
    _Float16* W2_h  = (_Float16*)alloc((size_t)L * D4 * D * 2);

    auto cvt = [&](const float* s, _Float16* d, size_t n) {
        const int n4 = (int)(n / 4);
        k_cvt_f16<<<(n4 + TBLK - 1) / TBLK, TBLK, 0, stream>>>(s, d, n4);
    };
    cvt(emb, emb_h, (size_t)V * D);
    cvt(Wq, Wq_h, (size_t)L * D * D);
    cvt(Wk, Wk_h, (size_t)L * D * D);
    cvt(Wv, Wv_h, (size_t)L * D * D);
    cvt(Wo, Wo_h, (size_t)L * D * D);
    cvt(W1, W1_h, (size_t)L * D4 * D);
    cvt(W2, W2_h, (size_t)L * D4 * D);

    // GEMM wrappers
    auto gemm_wt_f32 = [&](const _Float16* A, int lda, const _Float16* B, int ldb,
                           float* C, int ldc, int M, int N, int K,
                           float alpha, const float* bias, int causal) {
        dim3 grid(N / BN, M / BM);
        k_gemm_wmma<1, false, false><<<grid, TBLK, 0, stream>>>(
            A, lda, B, ldb, C, ldc, M, N, K, alpha, bias, causal);
    };
    auto gemm_wt_f16 = [&](const _Float16* A, int lda, const _Float16* B, int ldb,
                           _Float16* C, int ldc, int M, int N, int K,
                           const float* bias) {
        dim3 grid(N / BN, M / BM);
        k_gemm_wmma<1, false, true><<<grid, TBLK, 0, stream>>>(
            A, lda, B, ldb, C, ldc, M, N, K, 1.0f, bias, 0);
    };
    auto gemm_nn_f16 = [&](const _Float16* A, int lda, const _Float16* B, int ldb,
                           _Float16* C, int ldc, int M, int N, int K) {
        dim3 grid(N / BN, M / BM);
        k_gemm_wmma<0, false, true><<<grid, TBLK, 0, stream>>>(
            A, lda, B, ldb, C, ldc, M, N, K, 1.0f, nullptr, 0);
    };

    // token embedding + sinusoidal PE
    k_embed<<<T, TBLK, 0, stream>>>(tok, emb, x, T, D);

    const float iscale = 0.125f;  // 1/sqrt(HD=64)
    const int nTD  = T * D;
    const int nTD4 = T * D4;

    for (int l = 0; l < L; ++l) {
        const size_t wOff  = (size_t)l * D * D;
        const size_t w1Off = (size_t)l * D4 * D;

        k_layernorm<<<T, TBLK, 0, stream>>>(x, ln1_g + l * D, ln1_b + l * D, xn_h, D);

        // QKV projections -> f16 activations
        gemm_wt_f16(xn_h, D, Wq_h + wOff, D, qh, D, T, D, D, bq + (size_t)l * D);
        gemm_wt_f16(xn_h, D, Wk_h + wOff, D, kh, D, T, D, D, bk + (size_t)l * D);
        gemm_wt_f16(xn_h, D, Wv_h + wOff, D, vh, D, T, D, D, bv + (size_t)l * D);

        for (int h = 0; h < H; ++h) {
            // S = (Q_h K_h^T)/8 fp32; fully-masked 64x64 tiles skipped
            gemm_wt_f32(qh + h * HD, D, kh + h * HD, D, scores, T,
                        T, T, HD, iscale, nullptr, 1);
            k_softmax_causal<<<T, TBLK, 0, stream>>>(scores, prob_h, T);
            // ctx_h = P V_h into concat slot h
            gemm_nn_f16(prob_h, T, vh + h * HD, D, ctx_h + h * HD, D, T, HD, T);
        }

        // output projection + residual
        gemm_wt_f32(ctx_h, D, Wo_h + wOff, D, tmp, D, T, D, D,
                    1.0f, bo + (size_t)l * D, 0);
        k_add_inplace<<<(nTD + TBLK - 1) / TBLK, TBLK, 0, stream>>>(x, tmp, nTD);

        // MLP
        k_layernorm<<<T, TBLK, 0, stream>>>(x, ln2_g + l * D, ln2_b + l * D, xn_h, D);
        gemm_wt_f32(xn_h, D, W1_h + w1Off, D, hbuf, D4, T, D4, D,
                    1.0f, b1 + (size_t)l * D4, 0);
        k_gelu_h<<<(nTD4 + TBLK - 1) / TBLK, TBLK, 0, stream>>>(hbuf, hh, nTD4);
        gemm_wt_f32(hh, D4, W2_h + w1Off, D4, tmp, D, T, D, D4,
                    1.0f, b2 + (size_t)l * D, 0);
        k_add_inplace<<<(nTD + TBLK - 1) / TBLK, TBLK, 0, stream>>>(x, tmp, nTD);
    }

    // final LN + tied lm_head (N = 50257 -> guarded instantiation)
    k_layernorm<<<T, TBLK, 0, stream>>>(x, lnf_g, lnf_b, xn_h, D);
    {
        dim3 grid((V + BN - 1) / BN, T / BM);
        k_gemm_wmma<1, true, false><<<grid, TBLK, 0, stream>>>(
            xn_h, D, emb_h, D, logits, V, T, V, D, 1.0f, lm_bias, 0);
    }
}
